// PointerNet_38517266711124
// MI455X (gfx1250) — matrix operand order repaired and checked by reading, not compiled, not run on hardware
//
#include <hip/hip_runtime.h>
#include <hip/hip_bf16.h>

// ---------------------------------------------------------------------------
// PointerNet attention for MI455X (gfx1250, wave32).
//   Phase 1: two GEMM+bias via v_wmma_f32_16x16x32_f16 (f16 mul, f32 acc),
//            B tile staged once per block in LDS (kills 8x redundant loads).
//   Phase 2: tanh(q+h)·v + masked softmax, trans-op bound (v_exp/v_rcp).
// Sizes (fixed by the reference): B=4, DEC=256, ENC=256, E=512, D=1024, H=512.
// Workspace: q_trans (1024x512 f32) + h_trans (1024x512 f32) = 4 MB.
// ---------------------------------------------------------------------------

#define PN_B    4
#define PN_DEC  256
#define PN_ENC  256
#define PN_E    512
#define PN_D    1024
#define PN_H    512

typedef __attribute__((ext_vector_type(16))) _Float16 v16h;
typedef __attribute__((ext_vector_type(8)))  float    v8f;

// ---------------------------------------------------------------------------
// GEMM + bias:  out[M,N] = A[M,K] * W[K,N] + bias[N]   (f32 in memory, f16
// WMMA multiplicands, f32 accumulators). 256 threads = 8 wave32 per block;
// each wave owns one 16x16 output tile; all waves share one 16-wide N tile.
// Per K-chunk (32), the 32x16 B tile is converted f32->f16 and staged in LDS
// in *fragment order* sB[lane][j], so each lane's WMMA operand is a single
// contiguous 32-byte LDS vector read.
// ---------------------------------------------------------------------------
__global__ __launch_bounds__(256)
void pn_wmma_gemm_bias(const float* __restrict__ A,
                       const float* __restrict__ W,
                       const float* __restrict__ bias,
                       float* __restrict__ out,
                       int M, int N, int K) {
  // B fragment staging buffer: 32 lanes x 16 f16 = 1 KB.
  __shared__ __align__(32) _Float16 sB[32 * 16];

  const int tid  = threadIdx.x;
  const int lane = tid & 31;
  const int wave = tid >> 5;
  const int lo   = lane & 15;     // row (A/C) or column (B) within tile
  const int hi   = lane >> 4;     // half-wave selector for K striping

  const int tn = blockIdx.x * 16;                 // N tile origin (shared)
  const int tm = (blockIdx.y * 8 + wave) * 16;    // M tile origin (per wave)

  // Staging role: pseudo-lane sl covers fragment lanes, wave id covers j.
  // ISA B layout (32x16 f16): lane = column N; lanes 0-15 carry K {0..7,16..23},
  // lanes 16-31 carry K {8..15,24..31} (same order as A).
  const int sl   = tid & 31;          // fragment lane being staged
  const int sj   = (tid >> 5) << 1;   // j, j+1 staged by this thread
  const int scol = tn + (sl & 15);    // global column
  const int skof = ((sj < 8) ? 0 : 16) + (sl >> 4) * 8 + (sj & 7); // K offset within chunk for b[sj]

  v8f c = {};  // 16x16 f32 accumulator tile

  const float* arow = A + (size_t)(tm + lo) * K;  // this lane's A row

  for (int k0 = 0; k0 < K; k0 += 32) {
    // --- Stage B tile (all 256 threads; coalesced 64B along N per row).
    __syncthreads();  // previous chunk's fragment reads complete
    {
      const int kr = k0 + skof;
      sB[sl * 16 + sj]     = (_Float16)W[(size_t)kr * N + scol];
      sB[sl * 16 + sj + 1] = (_Float16)W[(size_t)(kr + 1) * N + scol];
    }
    __syncthreads();

    // --- A fragment: 16x32 f16. ISA layout: lanes 0-15 hold K {0..7,16..23},
    //     lanes 16-31 hold K {8..15,24..31}, 2 packed f16 per VGPR.
    v16h a;
#pragma unroll
    for (int j = 0; j < 8; ++j) {
      const int k = ((j < 4) ? 0 : 16) + hi * 8 + 2 * (j & 3);
      const float2 f = *(const float2*)(arow + k0 + k);
      a[2 * j]     = (_Float16)f.x;
      a[2 * j + 1] = (_Float16)f.y;
    }

    // --- B fragment: one contiguous 32-byte LDS vector per lane.
    const v16h b = *(const v16h*)(sB + lane * 16);

    // D = A*B + C  (8 args: neg_a, A, neg_b, B, c_mod, C, reuse_a, reuse_b)
    c = __builtin_amdgcn_wmma_f32_16x16x32_f16(false, a, false, b,
                                               (short)0, c, false, false);
  }

  // --- Epilogue: C/D layout -> VGPR r holds M = r (lanes 0-15) / r+8 (16-31),
  //     N = lane&15. Add bias and store f32.
  const int n  = tn + lo;
  const float bv = bias[n];
#pragma unroll
  for (int r = 0; r < 8; ++r) {
    const int m = tm + r + hi * 8;
    out[(size_t)m * N + n] = c[r] + bv;
  }
}

// ---------------------------------------------------------------------------
// Fast tanh: (e^{2x}-1)/(e^{2x}+1) via v_exp_f32 + v_rcp_f32 (2 trans ops).
// Clamp keeps e^{2x} finite; |x|>9 is already tanh==±1 to f32 precision.
// ---------------------------------------------------------------------------
__device__ __forceinline__ float pn_fast_tanh(float x) {
  const float xc = fminf(fmaxf(x, -9.0f), 9.0f);
  const float e2 = __builtin_amdgcn_exp2f(xc * 2.8853900817779268f); // 2*log2(e)
  return (e2 - 1.0f) * __builtin_amdgcn_rcpf(e2 + 1.0f);
}

// ---------------------------------------------------------------------------
// Phase 2: per (b,d) block of 256 threads (8 wave32), one thread per e.
//   score[e] = sum_h tanh(q[b,e,h] + h[b,d,h]) * v[h] + b2
//   out[e]   = exp(score - max) * mask[e] / sum   (mask AFTER exp, as in ref)
// h-row and v are broadcast from LDS (ds_load_b128); q streams per-lane with
// float4 loads (8 successive loads hit each 128B line -> high WGP$ reuse).
// ---------------------------------------------------------------------------
__global__ __launch_bounds__(256)
void pn_attn_softmax(const float* __restrict__ q_trans,   // [B*ENC, H]
                     const float* __restrict__ h_trans,   // [B*DEC, H]
                     const float* __restrict__ v,         // [H]
                     const float* __restrict__ b2,        // [1]
                     const int*   __restrict__ mask,      // [B, ENC]
                     float* __restrict__ out) {           // [B, DEC, ENC]
  __shared__ __align__(16) float s_h[PN_H];
  __shared__ __align__(16) float s_v[PN_H];
  __shared__ float s_red[256];

  const int b   = blockIdx.x / PN_DEC;
  const int d   = blockIdx.x % PN_DEC;
  const int e   = threadIdx.x;

  // Cooperative, coalesced load of the (b,d) hidden row and v into LDS.
  const float* hrow = h_trans + (size_t)(b * PN_DEC + d) * PN_H;
  for (int i = threadIdx.x; i < PN_H; i += 256) {
    s_h[i] = hrow[i];
    s_v[i] = v[i];
  }
  __syncthreads();

  // Main reduction over H: 512 fast-tanh + FMA per thread.
  const float* qrow = q_trans + (size_t)(b * PN_ENC + e) * PN_H;
  float acc = 0.0f;
#pragma unroll 4
  for (int h = 0; h < PN_H; h += 4) {
    const float4 q4 = *(const float4*)(qrow + h);
    const float4 h4 = *(const float4*)(s_h + h);
    const float4 v4 = *(const float4*)(s_v + h);
    acc += pn_fast_tanh(q4.x + h4.x) * v4.x;
    acc += pn_fast_tanh(q4.y + h4.y) * v4.y;
    acc += pn_fast_tanh(q4.z + h4.z) * v4.z;
    acc += pn_fast_tanh(q4.w + h4.w) * v4.w;
  }
  const float score = acc + b2[0];

  // Block-wide max (over all e, BEFORE masking — faithful to reference).
  s_red[e] = score;
  __syncthreads();
  for (int s = 128; s > 0; s >>= 1) {
    if (e < s) s_red[e] = fmaxf(s_red[e], s_red[e + s]);
    __syncthreads();
  }
  const float mx = s_red[0];
  __syncthreads();

  // exp, then mask, then normalize.
  const float m  = (float)mask[b * PN_ENC + e];
  const float ex = __builtin_amdgcn_exp2f((score - mx) * 1.4426950408889634f) * m;
  s_red[e] = ex;
  __syncthreads();
  for (int s = 128; s > 0; s >>= 1) {
    if (e < s) s_red[e] = s_red[e] + s_red[e + s];
    __syncthreads();
  }
  const float inv = __builtin_amdgcn_rcpf(s_red[0]);
  out[(size_t)(b * PN_DEC + d) * PN_ENC + e] = ex * inv;
}

// ---------------------------------------------------------------------------
// Launch. Input order (setup_inputs dict): query_embed, mask, decoder_states,
// W1, b1, Wh, bh, v, b2. Workspace: 4 MB f32 (q_trans | h_trans).
// ---------------------------------------------------------------------------
extern "C" void kernel_launch(void* const* d_in, const int* in_sizes, int n_in,
                              void* d_out, int out_size, void* d_ws, size_t ws_size,
                              hipStream_t stream) {
  (void)in_sizes; (void)n_in; (void)out_size; (void)ws_size;

  const float* query_embed    = (const float*)d_in[0];
  const int*   mask           = (const int*)  d_in[1];
  const float* decoder_states = (const float*)d_in[2];
  const float* W1             = (const float*)d_in[3];
  const float* b1             = (const float*)d_in[4];
  const float* Wh             = (const float*)d_in[5];
  const float* bh             = (const float*)d_in[6];
  const float* v              = (const float*)d_in[7];
  const float* b2             = (const float*)d_in[8];
  float*       out            = (float*)d_out;

  float* q_trans = (float*)d_ws;                              // [1024, 512]
  float* h_trans = q_trans + (size_t)(PN_B * PN_ENC) * PN_H;  // [1024, 512]

  // GEMM 1: [B*ENC, E] x [E, H]  -> q_trans
  {
    dim3 grid(PN_H / 16, (PN_B * PN_ENC) / (16 * 8));
    pn_wmma_gemm_bias<<<grid, 256, 0, stream>>>(query_embed, W1, b1, q_trans,
                                                PN_B * PN_ENC, PN_H, PN_E);
  }
  // GEMM 2: [B*DEC, D] x [D, H]  -> h_trans
  {
    dim3 grid(PN_H / 16, (PN_B * PN_DEC) / (16 * 8));
    pn_wmma_gemm_bias<<<grid, 256, 0, stream>>>(decoder_states, Wh, bh, h_trans,
                                                PN_B * PN_DEC, PN_H, PN_D);
  }
  // Phase 2: tanh-attention + masked softmax.
  {
    dim3 grid(PN_B * PN_DEC);
    pn_attn_softmax<<<grid, 256, 0, stream>>>(q_trans, h_trans, v, b2, mask, out);
  }
}
